// PosetLoss_G2G_50251117363899
// MI455X (gfx1250) — matrix operand order repaired and checked by reading, not compiled, not run on hardware
//
#include <hip/hip_runtime.h>
#include <hip/hip_bf16.h>
#include <math.h>

// ---------------------------------------------------------------------------
// PosetLoss G2G for MI455X (gfx1250, wave32, WMMA + async-to-LDS staging)
// pool -> bf16 Gram via v_wmma_f32_16x16x32_bf16 (B panel double-buffered in
// LDS via global_load_async_to_lds_b128 / ASYNCcnt) with fused
// exp/rowsum/top-4 epilogue -> partial merge -> mean reduce.
// ---------------------------------------------------------------------------

typedef __attribute__((ext_vector_type(16))) __bf16 bf16x16;
typedef __attribute__((ext_vector_type(8)))  __bf16 bf16x8;
typedef __attribute__((ext_vector_type(8)))  float  f32x8;

// Problem shape fixed by setup_inputs()
constexpr int S = 32768;            // samples
constexpr int D = 512;              // feature dim
constexpr int I = 8;                // instances per group
constexpr int T = 8;                // tasks (cameras)
constexpr int G = S / I;            // 4096 groups
constexpr int KTOP = T / 2;         // 4 (top-K)
static_assert(KTOP == 4, "top-4 logic assumes task_num == 8");

constexpr int COLSPLIT = 8;         // column partitions of the Gram sweep
constexpr int PANEL    = 64;        // rows per block
constexpr int CHUNK    = 64;        // columns per inner tile
constexpr int SPAN     = G / COLSPLIT;   // 512 columns per block
constexpr int NCHUNK   = SPAN / CHUNK;   // 8
constexpr int BROW     = D + 8;     // padded LDS row (520 bf16 = 1040B = 260 dw
                                    // -> lane l15 starts at bank 4*l15: no conflicts)

__device__ __forceinline__ void top4_insert(float e, float& t0, float& t1,
                                            float& t2, float& t3) {
  if (e > t3) {
    if (e > t1) {
      if (e > t0) { t3 = t2; t2 = t1; t1 = t0; t0 = e; }
      else        { t3 = t2; t2 = t1; t1 = e; }
    } else {
      if (e > t2) { t3 = t2; t2 = e; }
      else        { t3 = e; }
    }
  }
}

// Issue one chunk's B panel copy: 64 rows x 1024B, 16 async b128 ops per
// thread (16 per wave on ASYNCcnt). LDS rows padded to BROW elements.
__device__ __forceinline__ void async_copy_panel(const __bf16* __restrict__ Fb,
                                                 int j0, __bf16 (*buf)[BROW],
                                                 int tid) {
  const int rr = tid >> 2;          // row 0..63
  const int qq = tid & 3;           // quarter of the row
  const char* g = (const char*)(Fb + (size_t)j0 * D) + rr * (D * 2) + qq * 256;
  const unsigned lbase = (unsigned)(size_t)(&buf[rr][0]) + qq * 256;
  #pragma unroll
  for (int i = 0; i < 16; ++i) {
    const unsigned long long ga = (unsigned long long)(size_t)(g + i * 16);
    const unsigned la = lbase + i * 16;
    asm volatile("global_load_async_to_lds_b128 %0, %1, off"
                 :: "v"(la), "v"(ga) : "memory");
  }
}

// ---------------------------------------------------------------------------
// Kernel 1: group mean-pool, bf16 feature matrix, f32 squared norms, group cam
// ---------------------------------------------------------------------------
__global__ void __launch_bounds__(256)
poset_pool_kernel(const float* __restrict__ x, const int* __restrict__ camid,
                  __bf16* __restrict__ Fb, float* __restrict__ sqv,
                  int* __restrict__ gcam) {
  const int g = blockIdx.x;
  const int t = threadIdx.x;
  const int c = t * 2;
  const float* row0 = x + (size_t)g * I * D;

  float ax = 0.0f, ay = 0.0f;
  #pragma unroll
  for (int r = 0; r < I; ++r) {
    const float2 v = *(const float2*)(row0 + r * D + c);
    ax += v.x; ay += v.y;
  }
  ax *= (1.0f / I); ay *= (1.0f / I);

  __bf16* dst = Fb + (size_t)g * D + c;
  dst[0] = (__bf16)ax;
  dst[1] = (__bf16)ay;

  __shared__ float red[256];
  red[t] = ax * ax + ay * ay;        // f32 squared norm of group mean
  __syncthreads();
  for (int w = 128; w > 0; w >>= 1) {
    if (t < w) red[t] += red[t + w];
    __syncthreads();
  }
  if (t == 0) { sqv[g] = red[0]; gcam[g] = camid[g * I]; }
}

// ---------------------------------------------------------------------------
// Kernel 2: 64x64 Gram tiles via WMMA, B panels async-staged in LDS,
// fused exp/rowsum/top-4 epilogue. Block = 8 waves (4 row x 2 col tiles).
// ---------------------------------------------------------------------------
__global__ void __launch_bounds__(256)
poset_gram_kernel(const __bf16* __restrict__ Fb, const float* __restrict__ sqv,
                  const int* __restrict__ gcam, float* __restrict__ psum,
                  float4* __restrict__ ptop) {
  __shared__ __bf16 Bbuf[2][CHUNK][BROW];       // 2 x 65KB double buffer
  __shared__ float  tile[PANEL][CHUNK + 1];
  __shared__ float  sh_sqj[CHUNK];
  __shared__ int    sh_cam[CHUNK];
  __shared__ float  shSum[256];
  __shared__ float  shTop[256][4];

  const int cs  = blockIdx.x;            // column split 0..7
  const int i0  = blockIdx.y * PANEL;    // row base
  const int j00 = cs * SPAN;             // column base of sweep
  const int tid = threadIdx.x;

  // wave-level matrix coordinates
  const int lane = tid & 31;
  const int wv   = tid >> 5;
  const int wr   = wv >> 1;              // row tile 0..3
  const int wc   = wv & 1;               // col pair 0..1
  const int half = lane >> 4;
  const int l15  = lane & 15;

  // scan-phase coordinates: 4 threads per row, 16 cols each
  const int r = tid >> 2;
  const int q = tid & 3;
  const float sqi  = sqv[i0 + r];
  const int   cami = gcam[i0 + r];

  float rsum = 0.0f, t0 = 0.0f, t1 = 0.0f, t2 = 0.0f, t3 = 0.0f;

  // A fragment base: lane<16 holds K {0..15}, lane>=16 {16..31}
  const __bf16* Arow = Fb + (size_t)(i0 + wr * 16 + l15) * D + half * 16;

  // Prefetch chunk 0's B panel.
  async_copy_panel(Fb, j00, Bbuf[0], tid);

  for (int ch = 0; ch < NCHUNK; ++ch) {
    const int j0 = j00 + ch * CHUNK;
    if (tid < CHUNK) { sh_sqj[tid] = sqv[j0 + tid]; sh_cam[tid] = gcam[j0 + tid]; }

    // Kick off next chunk's panel, then wait only for the current one.
    if (ch + 1 < NCHUNK) {
      async_copy_panel(Fb, j0 + CHUNK, Bbuf[(ch + 1) & 1], tid);
      asm volatile("s_wait_asynccnt 0x10" ::: "memory");  // current 16 landed
    } else {
      asm volatile("s_wait_asynccnt 0x0" ::: "memory");
    }
    __syncthreads();   // all waves' async copies for chunk ch now visible

    // B fragment base in LDS: lane holds column N=l15, K contiguous half*16..
    const __bf16* Bl0 = &Bbuf[ch & 1][wc * 32 + l15][half * 16];
    const __bf16* Bl1 = Bl0 + 16 * BROW;

    f32x8 acc0 = {0.f, 0.f, 0.f, 0.f, 0.f, 0.f, 0.f, 0.f};
    f32x8 acc1 = {0.f, 0.f, 0.f, 0.f, 0.f, 0.f, 0.f, 0.f};

    #pragma unroll
    for (int k0 = 0; k0 < D; k0 += 32) {
      bf16x16 a, b0, b1;
      const bf16x8 alo  = *(const bf16x8*)(Arow + k0);
      const bf16x8 ahi  = *(const bf16x8*)(Arow + k0 + 8);
      const bf16x8 b0lo = *(const bf16x8*)(Bl0 + k0);
      const bf16x8 b0hi = *(const bf16x8*)(Bl0 + k0 + 8);
      const bf16x8 b1lo = *(const bf16x8*)(Bl1 + k0);
      const bf16x8 b1hi = *(const bf16x8*)(Bl1 + k0 + 8);
      #pragma unroll
      for (int e = 0; e < 8; ++e) {
        a[e]  = alo[e];  a[e + 8]  = ahi[e];
        b0[e] = b0lo[e]; b0[e + 8] = b0hi[e];
        b1[e] = b1lo[e]; b1[e + 8] = b1hi[e];
      }
      acc0 = __builtin_amdgcn_wmma_f32_16x16x32_bf16(
          false, a, false, b0, (short)0, acc0, false, false);
      acc1 = __builtin_amdgcn_wmma_f32_16x16x32_bf16(
          false, a, false, b1, (short)0, acc1, false, false);
    }

    // C layout: VGPR rr -> row (half*8 + rr), col = l15
    #pragma unroll
    for (int rr = 0; rr < 8; ++rr) {
      const int mrow = wr * 16 + half * 8 + rr;
      tile[mrow][wc * 32 + l15]      = acc0[rr];
      tile[mrow][wc * 32 + 16 + l15] = acc1[rr];
    }
    __syncthreads();

    // fused epilogue: dist -> exp -> rowsum + cross-task top-4
    #pragma unroll
    for (int cc = 0; cc < 16; ++cc) {
      const int j = q * 16 + cc;
      float d = sqi + sh_sqj[j] - 2.0f * tile[r][j];
      d = fmaxf(d, 1e-12f);
      const float e = expf(-0.125f * sqrtf(d));   // -dist/(2*SIGMA), SIGMA=4
      rsum += e;
      if (sh_cam[j] != cami) top4_insert(e, t0, t1, t2, t3);
    }
    __syncthreads();   // tile + Bbuf[ch&1] free for reuse after this
  }

  // merge the 4 partials per row inside the block
  shSum[tid] = rsum;
  shTop[tid][0] = t0; shTop[tid][1] = t1; shTop[tid][2] = t2; shTop[tid][3] = t3;
  __syncthreads();
  if (tid < PANEL) {
    float s = 0.0f, m0 = 0.0f, m1 = 0.0f, m2 = 0.0f, m3 = 0.0f;
    #pragma unroll
    for (int qq = 0; qq < 4; ++qq) {
      const int idx = tid * 4 + qq;
      s += shSum[idx];
      #pragma unroll
      for (int k = 0; k < 4; ++k) top4_insert(shTop[idx][k], m0, m1, m2, m3);
    }
    const int gr = i0 + tid;
    psum[cs * G + gr] = s;
    ptop[cs * G + gr] = make_float4(m0, m1, m2, m3);
  }
}

// ---------------------------------------------------------------------------
// Kernel 3: merge COLSPLIT partials per row -> per-row loss term
// ---------------------------------------------------------------------------
__global__ void __launch_bounds__(256)
poset_merge_kernel(const float* __restrict__ psum, const float4* __restrict__ ptop,
                   float* __restrict__ rowterm) {
  const int g = blockIdx.x * blockDim.x + threadIdx.x;
  if (g >= G) return;
  float all = 0.0f, m0 = 0.0f, m1 = 0.0f, m2 = 0.0f, m3 = 0.0f;
  #pragma unroll
  for (int c = 0; c < COLSPLIT; ++c) {
    all += psum[c * G + g];
    const float4 v = ptop[c * G + g];
    top4_insert(v.x, m0, m1, m2, m3);
    top4_insert(v.y, m0, m1, m2, m3);
    top4_insert(v.z, m0, m1, m2, m3);
    top4_insert(v.w, m0, m1, m2, m3);
  }
  const float near = m0 + m1 + m2 + m3;
  rowterm[g] = logf(all) - logf(near);   // -log(near/all)
}

// ---------------------------------------------------------------------------
// Kernel 4: deterministic mean over G rows
// ---------------------------------------------------------------------------
__global__ void __launch_bounds__(256)
poset_final_kernel(const float* __restrict__ rowterm, float* __restrict__ out) {
  __shared__ float red[256];
  const int t = threadIdx.x;
  float s = 0.0f;
  for (int g = t; g < G; g += 256) s += rowterm[g];
  red[t] = s;
  __syncthreads();
  for (int w = 128; w > 0; w >>= 1) {
    if (t < w) red[t] += red[t + w];
    __syncthreads();
  }
  if (t == 0) out[0] = red[0] * (1.0f / G);
}

// ---------------------------------------------------------------------------
extern "C" void kernel_launch(void* const* d_in, const int* in_sizes, int n_in,
                              void* d_out, int out_size, void* d_ws, size_t ws_size,
                              hipStream_t stream) {
  (void)in_sizes; (void)n_in; (void)out_size; (void)ws_size;
  const float* x     = (const float*)d_in[0];   // batch_features [S,D] f32
  const int*   camid = (const int*)d_in[2];     // camid [S] i32
  // d_in[1] (labels), d_in[3] (num_instances), d_in[4] (task_num) unused.

  char* ws = (char*)d_ws;
  auto align256 = [](size_t v) { return (v + 255) & ~(size_t)255; };
  size_t off = 0;
  __bf16* Fb     = (__bf16*)(ws + off); off = align256(off + (size_t)G * D * sizeof(unsigned short));
  float*  sqv    = (float*)(ws + off);  off = align256(off + (size_t)G * sizeof(float));
  int*    gcam   = (int*)(ws + off);    off = align256(off + (size_t)G * sizeof(int));
  float*  psum   = (float*)(ws + off);  off = align256(off + (size_t)COLSPLIT * G * sizeof(float));
  float4* ptop   = (float4*)(ws + off); off = align256(off + (size_t)COLSPLIT * G * sizeof(float4));
  float* rowterm = (float*)(ws + off);  off = align256(off + (size_t)G * sizeof(float));

  poset_pool_kernel<<<G, 256, 0, stream>>>(x, camid, Fb, sqv, gcam);

  dim3 ggrid(COLSPLIT, G / PANEL);               // 8 x 64 = 512 blocks
  poset_gram_kernel<<<ggrid, 256, 0, stream>>>(Fb, sqv, gcam, psum, ptop);

  poset_merge_kernel<<<(G + 255) / 256, 256, 0, stream>>>(psum, ptop, rowterm);
  poset_final_kernel<<<1, 256, 0, stream>>>(rowterm, (float*)d_out);
}
